// RSLP_5188320494225
// MI455X (gfx1250) — matrix operand order repaired and checked by reading, not compiled, not run on hardware
//
#include <hip/hip_runtime.h>

typedef __bf16 bf16;
typedef bf16  v16bf __attribute__((ext_vector_type(16)));
typedef bf16  v8bf  __attribute__((ext_vector_type(8)));
typedef float v8f   __attribute__((ext_vector_type(8)));
typedef int   v4i   __attribute__((ext_vector_type(4)));

// address-space-qualified int4 for the async-LDS builtins (AS1 = global, AS3 = LDS)
typedef __attribute__((address_space(1))) v4i as1_v4i;
typedef __attribute__((address_space(3))) v4i as3_v4i;

#define DECAY_ 0.9f
#define B_     256
#define T_     32
#define N_     4096
#define DIN_   1024
#define DOUT_  256

// LDS tile geometry: 128 rows x 64 K-elems, padded row stride 72 elems (144B)
#define TK_      64
#define STRIDE_  72
#define BUFE_    (128 * STRIDE_)
#define NBUF_    4

#if __has_builtin(__builtin_amdgcn_global_load_async_to_lds_b128) && \
    __has_builtin(__builtin_amdgcn_s_wait_asynccnt)
#define HAVE_ASYNC_LDS 1
#else
#define HAVE_ASYNC_LDS 0
#endif

// ---------------- transpose + f32 -> bf16 : W[R][C] -> Wt[C][R] ----------------
__global__ void k_transpose_convert(const float* __restrict__ W, bf16* __restrict__ Wt,
                                    int R, int C) {
  __shared__ float tile[32][33];
  const int cB = blockIdx.x * 32, rB = blockIdx.y * 32;
  const int tx = threadIdx.x, ty = threadIdx.y;
#pragma unroll
  for (int i = 0; i < 32; i += 8)
    tile[ty + i][tx] = W[(size_t)(rB + ty + i) * C + (cB + tx)];
  __syncthreads();
#pragma unroll
  for (int i = 0; i < 32; i += 8)
    Wt[(size_t)(cB + ty + i) * R + (rB + tx)] = (bf16)tile[tx][ty + i];
}

// ---------------- f32 -> bf16 elementwise ----------------
__global__ void k_f32_to_bf16(const float* __restrict__ in, bf16* __restrict__ out) {
  int i = blockIdx.x * blockDim.x + threadIdx.x;
  out[i] = (bf16)in[i];
}

// ---------------- per-step state update ----------------
__global__ void k_update(const float* __restrict__ i_, const float* __restrict__ u_prev,
                         const float* __restrict__ r_prev, float* __restrict__ u_out,
                         bf16* __restrict__ a_out, int first) {
  int e = blockIdx.x * blockDim.x + threadIdx.x;   // 0 .. B*N-1
  int b = e >> 12;                                  // N_ == 4096
  int n = e & (N_ - 1);
  size_t o = (size_t)b * (T_ * N_) + n;             // strided slice in act [B,T,N]
  float drive = i_[e];
  float u;
  if (first) u = drive;
  else       u = DECAY_ * u_prev[o] + drive + r_prev[e];
  u_out[o] = u;
  a_out[e] = (bf16)fmaxf(u, 0.0f);
}

// ---------------- WMMA fragment loaders (LDS, padded row stride) ------
__device__ __forceinline__ v16bf frag_a(const bf16* lds, int rowBase, int lane) {
  int m  = rowBase + (lane & 15);
  int kh = lane >> 4;
  v8bf lo = *(const v8bf*)(lds + m * STRIDE_ + kh * 8);
  v8bf hi = *(const v8bf*)(lds + m * STRIDE_ + 16 + kh * 8);
  return __builtin_shufflevector(lo, hi, 0,1,2,3,4,5,6,7,8,9,10,11,12,13,14,15);
}
__device__ __forceinline__ v16bf frag_b(const bf16* lds, int colBase, int lane) {
  int n  = colBase + (lane & 15);
  int kh = lane >> 4;
  v8bf lo = *(const v8bf*)(lds + n * STRIDE_ + kh * 8);
  v8bf hi = *(const v8bf*)(lds + n * STRIDE_ + 16 + kh * 8);
  return __builtin_shufflevector(lo, hi, 0,1,2,3,4,5,6,7,8,9,10,11,12,13,14,15);
}

#if HAVE_ASYNC_LDS
__device__ __forceinline__ void cp_async16(const bf16* g, bf16* s) {
  __builtin_amdgcn_global_load_async_to_lds_b128((as1_v4i*)g, (as3_v4i*)s, 0, 0);
}
#endif

// ---------------- bf16 WMMA GEMM: D[m][j] = sum_k A[m][k] * Bt[j][k] (+bias) ----
// 256 threads = 8 waves (4x2), block tile 128x128, wave tile 32x64, K-step 64.
// Quad-buffered LDS, async staging 2 tiles ahead, one barrier per K-step.
__global__ __launch_bounds__(256)
void k_wmma_gemm(const bf16* __restrict__ A, const bf16* __restrict__ Bt, int K,
                 float* __restrict__ out0, int ld0, int ncols0,
                 float* __restrict__ out1, int ld1,
                 const float* __restrict__ bias) {
  __shared__ __align__(16) bf16 lds_a[NBUF_ * BUFE_];
  __shared__ __align__(16) bf16 lds_b[NBUF_ * BUFE_];

  const int tid  = threadIdx.x;
  const int lane = tid & 31;
  const int wave = tid >> 5;
  const int wm   = wave >> 1;                 // 0..3 : 32-row strip
  const int wn   = wave & 1;                  // 0..1 : 64-col strip
  const int mBase = blockIdx.y * 128;
  const int nBase = blockIdx.x * 128;

  // staging: 2 threads per tile row, 32 elems (64B = 4x b128) each
  const int r = tid >> 1;
  const int h = tid & 1;
  const bf16* gA = A  + (size_t)(mBase + r) * K + h * 32;
  const bf16* gB = Bt + (size_t)(nBase + r) * K + h * 32;
  bf16* sA = lds_a + r * STRIDE_ + h * 32;
  bf16* sB = lds_b + r * STRIDE_ + h * 32;

  v8f acc[2][4] = {};
  const int nkb = K / TK_;                    // >= 16 for all our shapes

#if HAVE_ASYNC_LDS
  auto stage = [&](int kb, int buf) {
    const bf16* pa = gA + (size_t)kb * TK_;
    const bf16* pb = gB + (size_t)kb * TK_;
    bf16* sa = sA + buf * BUFE_;
    bf16* sb = sB + buf * BUFE_;
#pragma unroll
    for (int c = 0; c < 4; ++c) {
      cp_async16(pa + c * 8, sa + c * 8);
      cp_async16(pb + c * 8, sb + c * 8);
    }
  };
  stage(0, 0);
  stage(1, 1);
#else
  uint4 ra[4], rb[4];
  auto issue = [&](int kb) {
    const bf16* pa = gA + (size_t)kb * TK_;
    const bf16* pb = gB + (size_t)kb * TK_;
#pragma unroll
    for (int c = 0; c < 4; ++c) {
      ra[c] = *(const uint4*)(pa + c * 8);
      rb[c] = *(const uint4*)(pb + c * 8);
    }
    __builtin_prefetch(pa + TK_, 0, 1);
    __builtin_prefetch(pb + TK_, 0, 1);
  };
  auto commit = [&](int buf) {
    bf16* sa = sA + buf * BUFE_;
    bf16* sb = sB + buf * BUFE_;
#pragma unroll
    for (int c = 0; c < 4; ++c) {
      *(uint4*)(sa + c * 8) = ra[c];
      *(uint4*)(sb + c * 8) = rb[c];
    }
  };
  issue(0);
#endif

  for (int kb = 0; kb < nkb; ++kb) {
    const int cur = kb & (NBUF_ - 1);
#if HAVE_ASYNC_LDS
    if (kb + 2 < nkb) {
      stage(kb + 2, (kb + 2) & (NBUF_ - 1));
      __builtin_amdgcn_s_wait_asynccnt(16);   // tile kb landed (in-order done)
    } else if (kb + 1 < nkb) {
      __builtin_amdgcn_s_wait_asynccnt(8);
    } else {
      __builtin_amdgcn_s_wait_asynccnt(0);
    }
    __syncthreads();
#else
    commit(cur);
    if (kb + 1 < nkb) issue(kb + 1);
    __syncthreads();
#endif

    const bf16* baseA = lds_a + cur * BUFE_;
    const bf16* baseB = lds_b + cur * BUFE_;
#pragma unroll
    for (int kc = 0; kc < 2; ++kc) {
      const bf16* la = baseA + kc * 32;
      const bf16* lb = baseB + kc * 32;
      // preload all fragments of this K-chunk, then issue the 8 WMMAs
      v16bf a_lo = frag_a(la, wm * 32,      lane);
      v16bf a_hi = frag_a(la, wm * 32 + 16, lane);
      v16bf b0 = frag_b(lb, wn * 64,      lane);
      v16bf b1 = frag_b(lb, wn * 64 + 16, lane);
      v16bf b2 = frag_b(lb, wn * 64 + 32, lane);
      v16bf b3 = frag_b(lb, wn * 64 + 48, lane);
      acc[0][0] = __builtin_amdgcn_wmma_f32_16x16x32_bf16(false, a_lo, false, b0, (short)0, acc[0][0], false, false);
      acc[1][0] = __builtin_amdgcn_wmma_f32_16x16x32_bf16(false, a_hi, false, b0, (short)0, acc[1][0], false, false);
      acc[0][1] = __builtin_amdgcn_wmma_f32_16x16x32_bf16(false, a_lo, false, b1, (short)0, acc[0][1], false, false);
      acc[1][1] = __builtin_amdgcn_wmma_f32_16x16x32_bf16(false, a_hi, false, b1, (short)0, acc[1][1], false, false);
      acc[0][2] = __builtin_amdgcn_wmma_f32_16x16x32_bf16(false, a_lo, false, b2, (short)0, acc[0][2], false, false);
      acc[1][2] = __builtin_amdgcn_wmma_f32_16x16x32_bf16(false, a_hi, false, b2, (short)0, acc[1][2], false, false);
      acc[0][3] = __builtin_amdgcn_wmma_f32_16x16x32_bf16(false, a_lo, false, b3, (short)0, acc[0][3], false, false);
      acc[1][3] = __builtin_amdgcn_wmma_f32_16x16x32_bf16(false, a_hi, false, b3, (short)0, acc[1][3], false, false);
    }
  }

  // epilogue: C/D 16x16 f32 layout -> VGPR j holds M = j + 8*(lane>>4), N = lane&15
  const int rowHalf = (lane >> 4) * 8;
  const int colLane = lane & 15;
#pragma unroll
  for (int mf = 0; mf < 2; ++mf) {
#pragma unroll
    for (int nf = 0; nf < 4; ++nf) {
      int col = nBase + wn * 64 + nf * 16 + colLane;
#pragma unroll
      for (int j = 0; j < 8; ++j) {
        int row = mBase + wm * 32 + mf * 16 + rowHalf + j;
        float v = acc[mf][nf][j];
        if (bias) v += bias[col];
        if (col < ncols0) out0[(size_t)row * ld0 + col] = v;
        else              out1[(size_t)row * ld1 + (col - ncols0)] = v;
      }
    }
  }
}

extern "C" void kernel_launch(void* const* d_in, const int* in_sizes, int n_in,
                              void* d_out, int out_size, void* d_ws, size_t ws_size,
                              hipStream_t stream) {
  const float* x    = (const float*)d_in[0];   // [B, DIN]
  const float* W_i  = (const float*)d_in[1];   // [DIN, N]
  const float* bvec = (const float*)d_in[2];   // [N]
  const float* W_r  = (const float*)d_in[3];   // [N, N]
  const float* W_f  = (const float*)d_in[4];   // [N, DOUT]

  char* ws = (char*)d_ws;
  size_t off = 0;
  bf16*  Wcat = (bf16*)(ws + off);  off += (size_t)(N_ + DOUT_) * N_ * sizeof(bf16); // [4352][4096]
  bf16*  WiT  = (bf16*)(ws + off);  off += (size_t)N_ * DIN_ * sizeof(bf16);         // [4096][1024]
  bf16*  xb   = (bf16*)(ws + off);  off += (size_t)B_ * DIN_ * sizeof(bf16);
  float* ibuf = (float*)(ws + off); off += (size_t)B_ * N_ * sizeof(float);
  bf16*  abuf = (bf16*)(ws + off);  off += (size_t)B_ * N_ * sizeof(bf16);
  float* rbuf = (float*)(ws + off); off += (size_t)B_ * N_ * sizeof(float);

  float* fs  = (float*)d_out;                  // [B, T, DOUT]
  float* act = fs + (size_t)B_ * T_ * DOUT_;   // [B, T, N]

  dim3 tb(32, 8);
  k_transpose_convert<<<dim3(N_ / 32, DIN_ / 32), tb, 0, stream>>>(W_i, WiT, DIN_, N_);
  k_transpose_convert<<<dim3(N_ / 32, N_ / 32), tb, 0, stream>>>(W_r, Wcat, N_, N_);
  k_transpose_convert<<<dim3(DOUT_ / 32, N_ / 32), tb, 0, stream>>>(
      W_f, Wcat + (size_t)N_ * N_, N_, DOUT_);
  k_f32_to_bf16<<<(B_ * DIN_) / 256, 256, 0, stream>>>(x, xb);
  // i_ = x @ W_i + b   (M=256, K=1024, Ncols=4096)
  k_wmma_gemm<<<dim3(N_ / 128, B_ / 128), 256, 0, stream>>>(
      xb, WiT, DIN_, ibuf, N_, N_, nullptr, 0, bvec);

  for (int t = 0; t < T_; ++t) {
    const float* uprev = act + (size_t)(t - 1) * N_;  // dead when t==0 (in-bounds)
    k_update<<<(B_ * N_) / 256, 256, 0, stream>>>(
        ibuf, uprev, rbuf, act + (size_t)t * N_, abuf, t == 0);
    // Fused [r_new | f] = a @ [W_r | W_f]   (M=256, K=4096, Ncols=4352)
    k_wmma_gemm<<<dim3((N_ + DOUT_) / 128, B_ / 128), 256, 0, stream>>>(
        abuf, Wcat, N_, rbuf, N_, N_, fs + (size_t)t * DOUT_, T_ * DOUT_, nullptr);
  }
}
